// MultiCrossAlign_head_atttrans_res1sepalign_38242388803697
// MI455X (gfx1250) — matrix operand order repaired and checked by reading, not compiled, not run on hardware
//
#include <hip/hip_runtime.h>

// ---------------------------------------------------------------------------
// MI455X (gfx1250) implementation of MultiCrossAlign.
// All GEMM-shaped work (3x3 convs as implicit GEMM, 1x1 projections, QK^T,
// PV, O-proj) goes through v_wmma_f32_16x16x32_f16 (f16 in, f32 accumulate).
//
// LDS tiles are stored in *fragment order* so each lane's 16-f16 WMMA operand
// is one aligned 32-byte chunk (ds_load_b128 x2) instead of 16 ds_load_u16.
// Operand layouts used here (wave32); A/B k order is this file's choice, not the ISA table's:
//   A 16x32 f16 : lane = m + 16*(k>>4),     elem = k & 15  (contiguous k)
//   B 32x16 f16 : lane = n + 16*(k>>4),     elem = k & 15  (contiguous k)
//   C/D 16x16 f32: vgpr r -> m = r + 8*(lane>>4), n = lane&15
// ---------------------------------------------------------------------------

typedef __attribute__((ext_vector_type(16))) _Float16 v16h;
typedef __attribute__((ext_vector_type(8)))  float    v8f;

// f16 offset of element (m,k) inside a fragment-ordered A tile (512 f16 / tile)
__device__ __forceinline__ int a_slot(int m, int k) {
  int lane = (m & 15) + ((k >> 4) << 4);
  int e = k & 15;
  return lane * 16 + e;
}
// f16 offset of element (k,n) inside a fragment-ordered B tile (512 f16 / tile)
__device__ __forceinline__ int b_slot(int k, int n) {
  int lane = (n & 15) + ((k >> 4) << 4);
  return lane * 16 + (k & 15);
}

// ---------------------------------------------------------------------------
// 3x3 conv (pad=1) as implicit GEMM.  Workgroup: 64 outch x 32 pixels.
// 8 waves, wave w -> tile (mt = w&3 outch-tile, nt = w>>2 pixel-tile).
// Optional concat input: channels [0,64) from in0, [64,Cin) from in1.
// act: 0=none 1=relu 2=sigmoid 3=leaky(0.1)
// ---------------------------------------------------------------------------
__global__ __launch_bounds__(256)
void conv3x3_wmma_kernel(const float* __restrict__ in0, const float* __restrict__ in1,
                         const float* __restrict__ wgt,  // [64, Cin, 3, 3]
                         const float* __restrict__ bias, float* __restrict__ out,
                         int Bn, int Cin, int Hin, int Win,
                         int Hout, int Wout, int stride, int act) {
  __shared__ v16h lds_a[4 * 32];   // 4 m-tiles, fragment order
  __shared__ v16h lds_b[2 * 32];   // 2 n-tiles, fragment order
  _Float16* sa = (_Float16*)lds_a;
  _Float16* sb = (_Float16*)lds_b;
  const int tid = threadIdx.x, lane = tid & 31, wave = tid >> 5;
  const int mt = wave & 3, nt = wave >> 2;
  const int npix = Bn * Hout * Wout;
  const int pixbase = blockIdx.x * 32;
  const int K = Cin * 9;
  v8f acc = {};
  for (int kc = 0; kc < K; kc += 32) {
    if (kc + 32 < K) __builtin_prefetch(&wgt[tid * 8 + kc + 32], 0, 1);
    // stage A (k-inner: both global read and LDS store contiguous per thread)
    {
      int mfull = tid >> 2;                 // 0..63
      int k0 = (tid & 3) * 8;
      _Float16* dst = sa + (mfull >> 4) * 512;
#pragma unroll
      for (int i = 0; i < 8; ++i) {
        int k = k0 + i;
        dst[a_slot(mfull, k)] = (_Float16)wgt[mfull * K + kc + k];
      }
    }
    // stage B (n-inner: global im2col reads contiguous across pixels)
    {
      int k = tid >> 3;                     // 0..31
      int n0 = (tid & 7) * 4;
      int kk = kc + k;
      int cin = kk / 9, rem = kk % 9;
      int dy = rem / 3, dx = rem % 3;
      const float* srcp = (cin < 64) ? in0 : in1;
      int c = (cin < 64) ? cin : cin - 64;
#pragma unroll
      for (int i = 0; i < 4; ++i) {
        int nfull = n0 + i;
        int pix = pixbase + nfull;
        float v = 0.f;
        if (pix < npix) {
          int ox = pix % Wout, oy = (pix / Wout) % Hout, b = pix / (Wout * Hout);
          int iy = oy * stride + dy - 1, ix = ox * stride + dx - 1;
          if (iy >= 0 && iy < Hin && ix >= 0 && ix < Win)
            v = srcp[(((size_t)b * 64 + c) * Hin + iy) * Win + ix];
        }
        sb[(nfull >> 4) * 512 + b_slot(k, nfull)] = (_Float16)v;
      }
    }
    __syncthreads();
    v16h a  = lds_a[mt * 32 + lane];
    v16h bf = lds_b[nt * 32 + lane];
    acc = __builtin_amdgcn_wmma_f32_16x16x32_f16(false, a, false, bf, (short)0, acc,
                                                 false, false);
    __syncthreads();
  }
  const int m0 = mt * 16 + ((lane >> 4) << 3);
  const int n = nt * 16 + (lane & 15);
  const int pix = pixbase + n;
  if (pix < npix) {
    int ox = pix % Wout, oy = (pix / Wout) % Hout, b = pix / (Wout * Hout);
#pragma unroll
    for (int r = 0; r < 8; ++r) {
      int m = m0 + r;
      float v = acc[r] + bias[m];
      if (act == 1)      v = fmaxf(v, 0.f);
      else if (act == 2) v = 1.f / (1.f + __expf(-v));
      else if (act == 3) v = (v > 0.f) ? v : 0.1f * v;
      out[(((size_t)b * 64 + m) * Hout + oy) * Wout + ox] = v;
    }
  }
}

// ---------------------------------------------------------------------------
// Per-pixel linear projection (1x1 conv): out[c] = sum_ci in[ci] * w[ci,c] + b[c]
// ---------------------------------------------------------------------------
__global__ __launch_bounds__(256)
void linproj_wmma_kernel(const float* __restrict__ in, const float* __restrict__ wgt,
                         const float* __restrict__ bias, float* __restrict__ out,
                         int npix, int HW) {
  __shared__ v16h lds_a[4 * 32];
  __shared__ v16h lds_b[2 * 32];
  _Float16* sa = (_Float16*)lds_a;
  _Float16* sb = (_Float16*)lds_b;
  const int tid = threadIdx.x, lane = tid & 31, wave = tid >> 5;
  const int mt = wave & 3, nt = wave >> 2;
  const int pixbase = blockIdx.x * 32;
  v8f acc = {};
  for (int kc = 0; kc < 64; kc += 32) {
    // A = W^T : A[m=co][k=ci]   (k-inner; LDS stores contiguous)
    {
      int mfull = tid >> 2;
      int k0 = (tid & 3) * 8;
      _Float16* dst = sa + (mfull >> 4) * 512;
#pragma unroll
      for (int i = 0; i < 8; ++i) {
        int k = k0 + i;
        dst[a_slot(mfull, k)] = (_Float16)wgt[(kc + k) * 64 + mfull];
      }
    }
    // B[k=ci][n=pix]  (n-inner; coalesced global reads)
    {
      int k = tid >> 3;
      int n0 = (tid & 7) * 4;
#pragma unroll
      for (int i = 0; i < 4; ++i) {
        int nfull = n0 + i;
        int pix = pixbase + nfull;
        float v = 0.f;
        if (pix < npix) {
          int b = pix / HW, p = pix % HW;
          v = in[((size_t)b * 64 + kc + k) * HW + p];
        }
        sb[(nfull >> 4) * 512 + b_slot(k, nfull)] = (_Float16)v;
      }
    }
    __syncthreads();
    v16h a  = lds_a[mt * 32 + lane];
    v16h bf = lds_b[nt * 32 + lane];
    acc = __builtin_amdgcn_wmma_f32_16x16x32_f16(false, a, false, bf, (short)0, acc,
                                                 false, false);
    __syncthreads();
  }
  const int m0 = mt * 16 + ((lane >> 4) << 3);
  const int n = nt * 16 + (lane & 15);
  const int pix = pixbase + n;
  if (pix < npix) {
    int b = pix / HW, p = pix % HW;
#pragma unroll
    for (int r = 0; r < 8; ++r) {
      int m = m0 + r;
      out[((size_t)b * 64 + m) * HW + p] = acc[r] + bias[m];
    }
  }
}

// ---------------------------------------------------------------------------
// Windowed cross-attention, one workgroup per 8x8 query window.
// q/k/v are pre-projected maps; OOB kv tokens take the k/v BIAS (0 @ W + b).
// All operand matrices live in LDS in fragment order; scores/head-outputs are
// stored per-tile [lane][8] so WMMA D writes out as one aligned v8f.
// Dynamic LDS: 224 KB (fits the 320 KB WGP LDS; one workgroup per WGP).
// ---------------------------------------------------------------------------
__global__ __launch_bounds__(256)
void win_cross_attn_kernel(const float* __restrict__ qproj, const float* __restrict__ kproj,
                           const float* __restrict__ vproj, const float* __restrict__ src,
                           const float* __restrict__ kb,    const float* __restrict__ vb,
                           const float* __restrict__ ow,    const float* __restrict__ ob,
                           float* __restrict__ out, int H, int W) {
  extern __shared__ v16h dynsmem[];
  v16h*  qfr  = dynsmem;                 // A-frags Q  [head][mt(4)][lane]   512  (16 KB)
  v16h*  kfr  = qfr + 512;               // B-frags K^T [head][nt(16)][lane] 2048 (64 KB)
  v16h*  vfr  = kfr + 2048;              // B-frags V  [head][kc(8)][lane]   1024 (32 KB)
  float* sc   = (float*)(vfr + 1024);    // scores [tile(64)][lane][r]       16384 f32 (64 KB)
  v16h*  spfr = (v16h*)(sc + 16384);     // A-frags P  [mt(4)][kc(8)][lane]  1024 (32 KB)
  float* oac  = (float*)(spfr + 1024);   // head out  [head][mt(4)][lane][r] 4096 f32 (16 KB)

  const int tid = threadIdx.x, lane = tid & 31, wave = tid >> 5;
  const int wx = blockIdx.x, wy = blockIdx.y, b = blockIdx.z;
  const int y0 = wy * 8, x0 = wx * 8;
  const size_t plane = (size_t)H * W;

  // zero the fragment regions that carry K-dim padding (qfr lanes 16..31,
  // kfr lanes 16..31); fills below only write the valid halves.
  {
    v16h z = {};
    for (int i = tid; i < 512 + 2048; i += 256)
      (i < 512 ? qfr[i] : kfr[i - 512]) = z;
  }
  __syncthreads();

  // ---- Q fill: thread -> (token t = tid>>2, head = tid&3), 16 feats -------
  {
    int t = tid >> 2, head = tid & 3;
    int y = y0 + (t >> 3), x = x0 + (t & 7);
    int mt = t >> 4, m = t & 15;
    _Float16* base = (_Float16*)(qfr + (head * 4 + mt) * 32);
#pragma unroll
    for (int kk = 0; kk < 16; ++kk) {
      float v = qproj[((size_t)b * 64 + head * 16 + kk) * plane + (size_t)y * W + x];
      base[m * 16 + kk] = (_Float16)v;
    }
  }
  // ---- K fill: thread -> kv token u = tid, all 64 feats --------------------
  {
    int u = tid;
    int y = y0 - 4 + (u >> 4), x = x0 - 4 + (u & 15);
    bool inb = (y >= 0 && y < H && x >= 0 && x < W);
    int nt = u >> 4, n = u & 15;
    for (int head = 0; head < 4; ++head) {
      _Float16* base = (_Float16*)(kfr + (head * 16 + nt) * 32) + n * 16;  // lanes 0..15
#pragma unroll
      for (int kk = 0; kk < 16; ++kk) {
        int c = head * 16 + kk;
        float v = inb ? kproj[((size_t)b * 64 + c) * plane + (size_t)y * W + x] : kb[c];
        base[kk] = (_Float16)v;
      }
    }
  }
  // ---- V fill: thread -> feat c = tid>>2, 64 kv tokens ---------------------
  {
    int c = tid >> 2, u0 = (tid & 3) * 64;
    int head = c >> 4, n = c & 15;
    for (int j = 0; j < 64; ++j) {
      int u = u0 + j;
      int y = y0 - 4 + (u >> 4), x = x0 - 4 + (u & 15);
      bool inb = (y >= 0 && y < H && x >= 0 && x < W);
      float v = inb ? vproj[((size_t)b * 64 + c) * plane + (size_t)y * W + x] : vb[c];
      int kc = u >> 5, kk = u & 31;
      ((_Float16*)(vfr + (head * 8 + kc) * 32))[(n + ((kk >> 4) << 4)) * 16 + (kk & 15)] =
          (_Float16)v;
    }
  }
  __syncthreads();

  for (int h = 0; h < 4; ++h) {
    // ---- scores: 64 tiles (4 m x 16 n), 8 per wave, one WMMA each ----------
#pragma unroll
    for (int j = 0; j < 8; ++j) {
      int tile = wave * 8 + j;
      int mt = tile >> 4, nt = tile & 15;
      v16h a  = qfr[(h * 4 + mt) * 32 + lane];
      v16h bf = kfr[(h * 16 + nt) * 32 + lane];
      v8f c0 = {};
      c0 = __builtin_amdgcn_wmma_f32_16x16x32_f16(false, a, false, bf, (short)0, c0,
                                                  false, false);
      v8f s;
#pragma unroll
      for (int r = 0; r < 8; ++r) s[r] = c0[r] * 0.25f;   // 1/sqrt(16)
      *(v8f*)(sc + (tile * 32 + lane) * 8) = s;
    }
    __syncthreads();
    // ---- softmax rows (threads 0..63), swizzled access ---------------------
    if (tid < 64) {
      int t = tid;
      int rowbase = (t >> 4) * 4096 + (((t & 15) >> 3) << 7) + (t & 7);
      float mx = -1e30f;
      for (int jt = 0; jt < 16; ++jt)
        for (int n2 = 0; n2 < 16; ++n2)
          mx = fmaxf(mx, sc[rowbase + jt * 256 + n2 * 8]);
      float s = 0.f;
      for (int jt = 0; jt < 16; ++jt)
        for (int n2 = 0; n2 < 16; ++n2) {
          int ad = rowbase + jt * 256 + n2 * 8;
          float e = __expf(sc[ad] - mx);
          sc[ad] = e;
          s += e;
        }
      float inv = 1.f / s;
      int mt = t >> 4, m = t & 15;
      for (int jt = 0; jt < 16; ++jt)
        for (int n2 = 0; n2 < 16; ++n2) {
          int jj = jt * 16 + n2;
          int kc = jj >> 5, kk = jj & 31;
          ((_Float16*)(spfr + (mt * 8 + kc) * 32))[a_slot(m, kk)] =
              (_Float16)(sc[rowbase + jt * 256 + n2 * 8] * inv);
        }
    }
    __syncthreads();
    // ---- O_h = P(64x256) x V_h(256x16): waves 0..3, 8 k-chunks -------------
    if (wave < 4) {
      int mt = wave;
      v8f c0 = {};
#pragma unroll
      for (int kc = 0; kc < 8; ++kc) {
        v16h a  = spfr[(mt * 8 + kc) * 32 + lane];
        v16h bf = vfr[(h * 8 + kc) * 32 + lane];
        c0 = __builtin_amdgcn_wmma_f32_16x16x32_f16(false, a, false, bf, (short)0, c0,
                                                    false, false);
      }
      *(v8f*)(oac + ((h * 4 + mt) * 32 + lane) * 8) = c0;
    }
    __syncthreads();
  }

  // ---- out = src + O @ Wo + bo : 16 tiles, 2 per wave ----------------------
  const int hh = lane >> 4;
#pragma unroll
  for (int j = 0; j < 2; ++j) {
    int tile = wave * 2 + j;
    int mt = tile >> 2, nt = tile & 3;
    v8f c0 = {};
    for (int kc = 0; kc < 64; kc += 32) {
      v16h a, bf;
#pragma unroll
      for (int e = 0; e < 16; ++e) {
        int k = hh * 16 + e;
        int c = kc + k;  // channel of O
        int ad = ((c >> 4) * 4 + mt) * 32 + (c & 15) + (((lane & 15) >> 3) << 4);
        a[e] = (_Float16)oac[ad * 8 + (lane & 7)];
      }
#pragma unroll
      for (int e = 0; e < 16; ++e) {
        int k = kc + hh * 16 + e;
        bf[e] = (_Float16)ow[k * 64 + nt * 16 + (lane & 15)];
      }
      c0 = __builtin_amdgcn_wmma_f32_16x16x32_f16(false, a, false, bf, (short)0, c0,
                                                  false, false);
    }
    int m0 = mt * 16 + hh * 8;
    int n = nt * 16 + (lane & 15);
#pragma unroll
    for (int r = 0; r < 8; ++r) {
      int t = m0 + r;
      int y = y0 + (t >> 3), x = x0 + (t & 7);
      size_t oi = ((size_t)b * 64 + n) * plane + (size_t)y * W + x;
      out[oi] = src[oi] + c0[r] + ob[n];
    }
  }
}

__global__ __launch_bounds__(256)
void mul_kernel(const float* __restrict__ a, const float* __restrict__ g,
                float* __restrict__ o, int n) {
  int i = blockIdx.x * blockDim.x + threadIdx.x;
  if (i < n) o[i] = a[i] * g[i];
}

__global__ __launch_bounds__(256)
void upsample2x_add_kernel(const float* __restrict__ prev, float* __restrict__ out,
                           int H, int W, int n) {
  int i = blockIdx.x * blockDim.x + threadIdx.x;
  if (i >= n) return;
  int x = i % W, y = (i / W) % H, bc = i / (W * H);
  out[i] += prev[((size_t)bc * (H / 2) + (y >> 1)) * (W / 2) + (x >> 1)];
}

// ---------------------------------------------------------------------------
// Host orchestration.
// ---------------------------------------------------------------------------
struct AlP { const float *qw, *qb, *kw, *kb, *vw, *vb, *ow, *ob; };

extern "C" void kernel_launch(void* const* d_in, const int* in_sizes, int n_in,
                              void* d_out, int out_size, void* d_ws, size_t ws_size,
                              hipStream_t stream) {
  (void)n_in; (void)out_size; (void)ws_size;
  const int B = 4, H = 128, W = 128;
  const size_t SZ0 = (size_t)B * 64 * H * W;        // 4,194,304
  const size_t SZ1 = SZ0 / 4, SZ2 = SZ0 / 16;
  float* out = (float*)d_out;
  float* ws  = (float*)d_ws;
  auto in = [&](int i) { return (const float*)d_in[i]; };

  // Detect the harness's flattening of setup_inputs():
  //  order A: python insertion order (x1 first, size 4194304)
  //  order B: jax tree_flatten (dict keys sorted -> params.align1[0].k.b first, size 64)
  const bool orderA = (in_sizes[0] == (int)SZ0);

  const float *x1, *x2, *x3;
  const float *cfw[3], *cfb[3], *gw[2][2], *gb[2][2], *pw[3][3], *pb[3][3];
  AlP al[2][3];
  if (orderA) {
    x1 = in(0); x2 = in(1); x3 = in(2);
    for (int i = 0; i < 3; ++i) { cfw[i] = in(3 + 2 * i); cfb[i] = in(4 + 2 * i); }
    for (int g = 0; g < 2; ++g)
      for (int j = 0; j < 2; ++j) { int t = 9 + g * 4 + j * 2; gw[g][j] = in(t); gb[g][j] = in(t + 1); }
    for (int p = 0; p < 3; ++p)
      for (int j = 0; j < 3; ++j) { int t = 17 + p * 6 + j * 2; pw[p][j] = in(t); pb[p][j] = in(t + 1); }
    for (int a = 0; a < 2; ++a)
      for (int i = 0; i < 3; ++i) {
        int t = 35 + a * 24 + i * 8;  // q.w q.b k.w k.b v.w v.b o.w o.b
        al[a][i] = { in(t), in(t + 1), in(t + 2), in(t + 3), in(t + 4), in(t + 5), in(t + 6), in(t + 7) };
      }
  } else {
    for (int a = 0; a < 2; ++a)
      for (int i = 0; i < 3; ++i) {
        int t = a * 24 + i * 8;       // sorted: k.b k.w o.b o.w q.b q.w v.b v.w
        al[a][i].kb = in(t + 0); al[a][i].kw = in(t + 1);
        al[a][i].ob = in(t + 2); al[a][i].ow = in(t + 3);
        al[a][i].qb = in(t + 4); al[a][i].qw = in(t + 5);
        al[a][i].vb = in(t + 6); al[a][i].vw = in(t + 7);
      }
    for (int i = 0; i < 3; ++i) { cfb[i] = in(48 + 2 * i); cfw[i] = in(49 + 2 * i); }
    for (int g = 0; g < 2; ++g)
      for (int j = 0; j < 2; ++j) { int t = 54 + g * 4 + j * 2; gb[g][j] = in(t); gw[g][j] = in(t + 1); }
    for (int p = 0; p < 3; ++p)
      for (int j = 0; j < 3; ++j) { int t = 62 + p * 6 + j * 2; pb[p][j] = in(t); pw[p][j] = in(t + 1); }
    x1 = in(80); x2 = in(81); x3 = in(82);
  }

  // Workspace layout (~155 MB of floats).
  float* f1   = ws;
  float* f3   = ws + SZ0;
  float* qbuf = ws + 2 * SZ0;   // gate-temp early, q-projection later
  float* kbuf = ws + 3 * SZ0;   // g1 early, k-projection later
  float* vbuf = ws + 4 * SZ0;   // g3 early, v-projection later
  const size_t PSZ = SZ0 + SZ1 + SZ2;
  float* pyr[3][3];
  for (int p = 0; p < 3; ++p) {
    float* bp = ws + 5 * SZ0 + p * PSZ;
    pyr[p][0] = bp; pyr[p][1] = bp + SZ0; pyr[p][2] = bp + SZ0 + SZ1;
  }
  float* aS1 = ws + 5 * SZ0 + 3 * PSZ;  // scale-1 cascade buffer (SZ1)
  float* aS2 = aS1 + SZ1;               // scale-2 cascade buffer (SZ2)

  float* f2  = out + SZ0;       // output slot 1
  float* x1g = out + 3 * SZ0;   // output slot 3
  float* x3g = out + 4 * SZ0;   // output slot 4

  auto conv = [&](const float* i0, const float* i1, const float* w, const float* bi,
                  float* o, int Cin, int Hin, int Win, int stride, int act) {
    int Hout = (Hin - 1) / stride + 1, Wout = (Win - 1) / stride + 1;
    int npix = B * Hout * Wout;
    conv3x3_wmma_kernel<<<dim3((npix + 31) / 32), dim3(256), 0, stream>>>(
        i0, i1, w, bi, o, B, Cin, Hin, Win, Hout, Wout, stride, act);
  };
  auto lin = [&](const float* ip, const float* w, const float* bi, float* o, int Hs) {
    int HW = Hs * Hs, npix = B * HW;
    linproj_wmma_kernel<<<dim3((npix + 31) / 32), dim3(256), 0, stream>>>(ip, w, bi, o, npix, HW);
  };
  const size_t ATTN_LDS = (512 + 2048 + 1024 + 1024) * sizeof(v16h) +
                          (16384 + 4096) * sizeof(float);   // 229376 B = 224 KB
  auto attn = [&](const float* qp, const float* kp, const float* vp, const float* srcp,
                  const AlP& p, float* o, int Hs) {
    dim3 g(Hs / 8, Hs / 8, B);
    win_cross_attn_kernel<<<g, dim3(256), ATTN_LDS, stream>>>(
        qp, kp, vp, srcp, p.kb, p.vb, p.ow, p.ob, o, Hs, Hs);
  };
  auto upadd = [&](const float* prev, float* o, int Hs) {
    int n = B * 64 * Hs * Hs;
    upsample2x_add_kernel<<<(n + 255) / 256, 256, 0, stream>>>(prev, o, Hs, Hs, n);
  };

  // ---- feature extraction + gating ----------------------------------------
  conv(x1, nullptr, cfw[0], cfb[0], f1, 64, H, W, 1, 0);
  conv(x2, nullptr, cfw[1], cfb[1], f2, 64, H, W, 1, 0);
  conv(x3, nullptr, cfw[2], cfb[2], f3, 64, H, W, 1, 0);
  conv(f1, f2, gw[0][0], gb[0][0], qbuf, 128, H, W, 1, 1);          // relu
  conv(qbuf, nullptr, gw[0][1], gb[0][1], kbuf, 64, H, W, 1, 2);    // sigmoid -> g1
  conv(f3, f2, gw[1][0], gb[1][0], qbuf, 128, H, W, 1, 1);
  conv(qbuf, nullptr, gw[1][1], gb[1][1], vbuf, 64, H, W, 1, 2);    // g3
  const int n0 = (int)SZ0;
  mul_kernel<<<(n0 + 255) / 256, 256, 0, stream>>>(f1, kbuf, x1g, n0);
  mul_kernel<<<(n0 + 255) / 256, 256, 0, stream>>>(f3, vbuf, x3g, n0);

  // ---- pyramids (leaky 0.1; strides 1,2,2) --------------------------------
  const float* psrc[3] = { x1g, f2, x3g };
  for (int p = 0; p < 3; ++p) {
    conv(psrc[p],   nullptr, pw[p][0], pb[p][0], pyr[p][0], 64, H,     W,     1, 3);
    conv(pyr[p][0], nullptr, pw[p][1], pb[p][1], pyr[p][1], 64, H,     W,     2, 3);
    conv(pyr[p][1], nullptr, pw[p][2], pb[p][2], pyr[p][2], 64, H / 2, W / 2, 2, 3);
  }

  // ---- coarse-to-fine cascaded cross-attention alignment ------------------
  float* afinal[2] = { out, out + 2 * SZ0 };   // a1 -> slot 0, a3 -> slot 2
  const int Hs[3] = { H, H / 2, H / 4 };
  for (int a = 0; a < 2; ++a) {
    float* const* srcl = (a == 0) ? pyr[0] : pyr[2];  // p1 or p3
    float* const* refl = pyr[1];                      // p2
    // scale 2 (32x32)
    lin(refl[2], al[a][2].qw, al[a][2].qb, qbuf, Hs[2]);
    lin(srcl[2], al[a][2].kw, al[a][2].kb, kbuf, Hs[2]);
    lin(srcl[2], al[a][2].vw, al[a][2].vb, vbuf, Hs[2]);
    attn(qbuf, kbuf, vbuf, srcl[2], al[a][2], aS2, Hs[2]);
    // scale 1 (64x64) + upsampled scale-2
    lin(refl[1], al[a][1].qw, al[a][1].qb, qbuf, Hs[1]);
    lin(srcl[1], al[a][1].kw, al[a][1].kb, kbuf, Hs[1]);
    lin(srcl[1], al[a][1].vw, al[a][1].vb, vbuf, Hs[1]);
    attn(qbuf, kbuf, vbuf, srcl[1], al[a][1], aS1, Hs[1]);
    upadd(aS2, aS1, Hs[1]);
    // scale 0 (128x128) + upsampled scale-1 -> final output
    lin(refl[0], al[a][0].qw, al[a][0].qb, qbuf, Hs[0]);
    lin(srcl[0], al[a][0].kw, al[a][0].kb, kbuf, Hs[0]);
    lin(srcl[0], al[a][0].vw, al[a][0].vb, vbuf, Hs[0]);
    attn(qbuf, kbuf, vbuf, srcl[0], al[a][0], afinal[a], Hs[0]);
    upadd(aS1, afinal[a], Hs[0]);
  }
}
